// RetentionCore_644245094445
// MI455X (gfx1250) — matrix operand order repaired and checked by reading, not compiled, not run on hardware
//
#include <hip/hip_runtime.h>

// ---------------------------------------------------------------------------
// RetentionCore for MI455X (gfx1250, wave32, WMMA):
//   out = (EMA-scan(x) @ W^T + b) + depthwise_conv5(x) + conv_b
// Pipeline:
//   K0: W fp32 -> bf16                       (2 MiB, one-time)
//   K1: chunk-local EMA end states           (x read #1)
//   K2: sequential chunk-carry combine       (tiny)
//   K3: exact EMA scan from carry-in -> S16(bf16) + fused 5-tap conv -> out
//       (x read #2; x stays in registers for both uses)
//   K4: bf16 WMMA GEMM 32768x1024x1024, epilogue out += acc + bias
// Memory-bound: ~0.77 GB total HBM traffic ~= 33 us @ 23.3 TB/s,
// GEMM compute ~= 14 us on the 16x16x32 bf16 WMMA path -> bf16 is right.
// ---------------------------------------------------------------------------

typedef __bf16 bf16;
typedef __attribute__((ext_vector_type(8)))  bf16  v8bf;
typedef __attribute__((ext_vector_type(16))) bf16  v16bf;
typedef __attribute__((ext_vector_type(8)))  float v8f;

constexpr int kB = 8, kT = 4096, kD = 1024;
constexpr int kRows = kB * kT;          // 32768 GEMM rows
constexpr int kNC = 32, kL = kT / kNC;  // 32 scan chunks of 128
constexpr int BM = 128, BN = 128, BK = 32;
constexpr int LSTR = 40;                // padded LDS row stride (bf16 elems)

union Frag16 { v16bf v; v8bf h[2]; };

__device__ __forceinline__ float sigmoidf_(float v) {
    return 1.0f / (1.0f + __expf(-v));
}

// --- K0: convert W (D x D, row-major [e][d]) to bf16 -----------------------
__global__ void w_to_bf16_kernel(const float* __restrict__ W,
                                 bf16* __restrict__ W16) {
    int i = blockIdx.x * blockDim.x + threadIdx.x;
    W16[i] = (bf16)W[i];
}

// --- K1: per-chunk local EMA end state (zero init) -------------------------
__global__ void ema_carry_kernel(const float* __restrict__ x,
                                 const float* __restrict__ decay_logit,
                                 float* __restrict__ carry) {
    int idx = blockIdx.x * blockDim.x + threadIdx.x;  // ((b*NC)+c)*D + d
    int d = idx % kD;
    int c = (idx / kD) % kNC;
    int b = idx / (kD * kNC);
    float a = sigmoidf_(decay_logit[d]);
    float oma = 1.0f - a;
    const float* xp = x + ((size_t)(b * kT + c * kL)) * kD + d;
    float s = 0.0f;
#pragma unroll 4
    for (int i = 0; i < kL; ++i) s = a * s + oma * xp[(size_t)i * kD];
    carry[idx] = s;
}

// --- K2: sequential combine across chunks; leaves carry-IN per chunk -------
__global__ void ema_combine_kernel(const float* __restrict__ decay_logit,
                                   float* __restrict__ carry) {
    int idx = blockIdx.x * blockDim.x + threadIdx.x;  // b*D + d
    int d = idx % kD;
    int b = idx / kD;
    float a = sigmoidf_(decay_logit[d]);
    float aL = a;
#pragma unroll
    for (int i = 0; i < 7; ++i) aL *= aL;  // a^128 by squaring
    float cf = 0.0f;
    for (int c = 0; c < kNC; ++c) {
        size_t o = ((size_t)(b * kNC + c)) * kD + d;
        float e = carry[o];
        carry[o] = cf;           // carry-in for chunk c
        cf = e + aL * cf;        // full state at end of chunk c
    }
}

// --- K3: exact scan from carry-in -> S16; fused depthwise conv5 -> out -----
__global__ void ema_scan_conv_kernel(const float* __restrict__ x,
                                     const float* __restrict__ decay_logit,
                                     const float* __restrict__ conv_w,
                                     const float* __restrict__ conv_b,
                                     const float* __restrict__ carry,
                                     bf16* __restrict__ S16,
                                     float* __restrict__ out) {
    int idx = blockIdx.x * blockDim.x + threadIdx.x;  // ((b*NC)+c)*D + d
    int d = idx % kD;
    int c = (idx / kD) % kNC;
    int b = idx / (kD * kNC);
    float a = sigmoidf_(decay_logit[d]);
    float oma = 1.0f - a;
    float w0 = conv_w[d * 5 + 0], w1 = conv_w[d * 5 + 1], w2 = conv_w[d * 5 + 2];
    float w3 = conv_w[d * 5 + 3], w4 = conv_w[d * 5 + 4];
    float cb = conv_b[d];

    const float* xb = x + (size_t)b * kT * kD + d;
    int t0 = c * kL;
    auto ld = [&](int t) -> float {
        return (t >= 0 && t < kT) ? xb[(size_t)t * kD] : 0.0f;
    };
    float xm2 = ld(t0 - 2), xm1 = ld(t0 - 1), xt = ld(t0), xp1 = ld(t0 + 1);
    float s = carry[idx];  // exact full carry-in for this chunk
#pragma unroll 4
    for (int i = 0; i < kL; ++i) {
        int t = t0 + i;
        float xp2 = ld(t + 2);
        s = a * s + oma * xt;
        size_t o = ((size_t)(b * kT + t)) * kD + d;
        S16[o] = (bf16)s;
        out[o] = w0 * xm2 + w1 * xm1 + w2 * xt + w3 * xp1 + w4 * xp2 + cb;
        xm2 = xm1; xm1 = xt; xt = xp1; xp1 = xp2;
    }
}

// --- K4: bf16 WMMA GEMM: out += S16 @ W16^T + bias -------------------------
// Block: 128x128 tile, 256 threads = 8 wave32 in 2(M) x 4(N); each wave owns a
// 64x32 subtile = 4x2 wmma 16x16 accumulators. Double-buffered LDS staging.
__global__ __launch_bounds__(256) void gemm_wmma_kernel(
    const bf16* __restrict__ S16, const bf16* __restrict__ W16,
    const float* __restrict__ bias, float* __restrict__ out) {
    __shared__ bf16 lA[2][BM * LSTR];  // A rows [m][k], padded stride
    __shared__ bf16 lB[2][BN * LSTR];  // B as [n][k] == W rows directly

    const int tid = threadIdx.x;
    const int lane = tid & 31;
    const int wave = tid >> 5;
    const int waveM = wave & 1;   // 2 waves over M (64 rows each)
    const int waveN = wave >> 1;  // 4 waves over N (32 cols each)
    const int rowBase = blockIdx.x * BM;
    const int colBase = blockIdx.y * BN;

    // Each thread stages 16 bf16 (32 B) of A and of B per k-step.
    const int lrow = tid >> 1;
    const int lhalf = (tid & 1) * 16;
    const bf16* gA = S16 + (size_t)(rowBase + lrow) * kD + lhalf;
    const bf16* gB = W16 + (size_t)(colBase + lrow) * kD + lhalf;

    v8f acc[4][2];
#pragma unroll
    for (int mt = 0; mt < 4; ++mt)
#pragma unroll
        for (int nt = 0; nt < 2; ++nt) acc[mt][nt] = (v8f)(0.0f);

    // Prime buffer 0.
    {
        uint4 a0 = *(const uint4*)(gA);
        uint4 a1 = *(const uint4*)(gA + 8);
        uint4 b0 = *(const uint4*)(gB);
        uint4 b1 = *(const uint4*)(gB + 8);
        *(uint4*)&lA[0][lrow * LSTR + lhalf] = a0;
        *(uint4*)&lA[0][lrow * LSTR + lhalf + 8] = a1;
        *(uint4*)&lB[0][lrow * LSTR + lhalf] = b0;
        *(uint4*)&lB[0][lrow * LSTR + lhalf + 8] = b1;
    }
    __syncthreads();

    constexpr int KT = kD / BK;  // 32 k-steps
    const int am_lo = (lane < 16) ? 0 : 8;   // A: K lanes 0-15 get {0..7,16..23}
    const int bk_lo = (lane < 16) ? 0 : 8;   // B: K lanes 0-15 get {0..7,16..23}

    for (int kt = 0; kt < KT; ++kt) {
        const int cur = kt & 1, nxt = cur ^ 1;

        uint4 na0{}, na1{}, nb0{}, nb1{};
        if (kt + 1 < KT) {
            const bf16* pA = gA + (kt + 1) * BK;
            const bf16* pB = gB + (kt + 1) * BK;
            na0 = *(const uint4*)pA;
            na1 = *(const uint4*)(pA + 8);
            nb0 = *(const uint4*)pB;
            nb1 = *(const uint4*)(pB + 8);
        }
        if (kt + 2 < KT)  // pull the tile after next toward L2/L0
            __builtin_prefetch(gA + (kt + 2) * BK, 0, 1);

        Frag16 fa[4], fb[2];
#pragma unroll
        for (int mt = 0; mt < 4; ++mt) {
            int m = waveM * 64 + mt * 16 + (lane & 15);
            fa[mt].h[0] = *(const v8bf*)&lA[cur][m * LSTR + am_lo];
            fa[mt].h[1] = *(const v8bf*)&lA[cur][m * LSTR + am_lo + 16];
        }
#pragma unroll
        for (int nt = 0; nt < 2; ++nt) {
            int n = waveN * 32 + nt * 16 + (lane & 15);
            fb[nt].h[0] = *(const v8bf*)&lB[cur][n * LSTR + bk_lo];
            fb[nt].h[1] = *(const v8bf*)&lB[cur][n * LSTR + bk_lo + 16];
        }

#pragma unroll
        for (int nt = 0; nt < 2; ++nt)
#pragma unroll
            for (int mt = 0; mt < 4; ++mt)
                acc[mt][nt] = __builtin_amdgcn_wmma_f32_16x16x32_bf16(
                    false, fa[mt].v, false, fb[nt].v,
                    (short)0, acc[mt][nt], false, false);

        if (kt + 1 < KT) {
            *(uint4*)&lA[nxt][lrow * LSTR + lhalf] = na0;
            *(uint4*)&lA[nxt][lrow * LSTR + lhalf + 8] = na1;
            *(uint4*)&lB[nxt][lrow * LSTR + lhalf] = nb0;
            *(uint4*)&lB[nxt][lrow * LSTR + lhalf + 8] = nb1;
            __syncthreads();
        }
    }

    // Epilogue: out already holds y_loc (+conv_b); add y_ret + linear bias.
    float be[2];
#pragma unroll
    for (int nt = 0; nt < 2; ++nt)
        be[nt] = bias[colBase + waveN * 32 + nt * 16 + (lane & 15)];
    const int mrow = ((lane >> 4) & 1) * 8;  // C layout: lanes>=16 hold M=8+j
#pragma unroll
    for (int mt = 0; mt < 4; ++mt)
#pragma unroll
        for (int nt = 0; nt < 2; ++nt) {
            int e = colBase + waveN * 32 + nt * 16 + (lane & 15);
#pragma unroll
            for (int j = 0; j < 8; ++j) {
                int r = rowBase + waveM * 64 + mt * 16 + mrow + j;
                size_t o = (size_t)r * kD + e;
                out[o] += acc[mt][nt][j] + be[nt];
            }
        }
}

// ---------------------------------------------------------------------------
extern "C" void kernel_launch(void* const* d_in, const int* in_sizes, int n_in,
                              void* d_out, int out_size, void* d_ws,
                              size_t ws_size, hipStream_t stream) {
    const float* x  = (const float*)d_in[0];
    const float* dl = (const float*)d_in[1];
    const float* W  = (const float*)d_in[2];
    const float* bv = (const float*)d_in[3];
    const float* cw = (const float*)d_in[4];
    const float* cb = (const float*)d_in[5];
    float* out = (float*)d_out;

    // Workspace layout (~70.3 MiB): S16 | W16 | carry
    char* ws = (char*)d_ws;
    bf16* S16 = (bf16*)ws;
    bf16* W16 = (bf16*)(ws + (size_t)kRows * kD * sizeof(bf16));
    float* carry = (float*)(ws + (size_t)kRows * kD * sizeof(bf16)
                               + (size_t)kD * kD * sizeof(bf16));

    w_to_bf16_kernel<<<(kD * kD) / 256, 256, 0, stream>>>(W, W16);
    ema_carry_kernel<<<(kB * kNC * kD) / 256, 256, 0, stream>>>(x, dl, carry);
    ema_combine_kernel<<<(kB * kD) / 256, 256, 0, stream>>>(dl, carry);
    ema_scan_conv_kernel<<<(kB * kNC * kD) / 256, 256, 0, stream>>>(
        x, dl, cw, cb, carry, S16, out);

    dim3 grid(kRows / BM, kD / BN);
    gemm_wmma_kernel<<<grid, 256, 0, stream>>>(S16, W16, bv, out);
}
